// LocalConv2dLayer_12472585027617
// MI455X (gfx1250) — matrix operand order repaired and checked by reading, not compiled, not run on hardware
//
#include <hip/hip_runtime.h>

// CDNA5 / gfx1250: wave32, WMMA 16x16x32 bf16 -> f32 accumulate.
// Double-buffered LDS, register-pipelined filter stream, scalarized
// (readfirstlane) im2col address math, v_perm bf16 pair-packing,
// coalesced float4 epilogue.

typedef __attribute__((ext_vector_type(16))) __bf16 v16bf;
typedef __attribute__((ext_vector_type(8)))  float  v8f;

#define N_BATCH 128
#define C_IN    64
#define HW      32
#define C_OUT   128
#define RF      576          // C_IN * 3 * 3
#define PIX     1024         // 32 * 32
#define KC      32           // K-step per WMMA chain
#define NSTEP   (RF / KC)    // 18
#define KSTR    40           // ushorts per LDS row (32 data + 8 pad; 80B stride)
#define PT      4            // pixels per workgroup (contiguous w in one row)
#define BUFU    (2 * PT * 128 * KSTR)   // ushorts per LDS buffer (lA + lB)

union Frag {
    uint4 q[2];
    v16bf v;
};

// round-to-nearest-even f32 -> bf16 sitting in the HIGH half of the u32
__device__ __forceinline__ unsigned rnd_bf(float f) {
    unsigned u = __float_as_uint(f);
    return u + 0x7FFFu + ((u >> 16) & 1u);
}
// pack bf16(a) into [15:0], bf16(b) into [31:16] with a single v_perm_b32
__device__ __forceinline__ unsigned pack2bf(float a, float b) {
    return __builtin_amdgcn_perm(rnd_bf(b), rnd_bf(a), 0x07060302u);
}

__global__ __launch_bounds__(512, 1)
void localconv2d_wmma_bf16(const float* __restrict__ x,
                           const float* __restrict__ filt,
                           const float* __restrict__ bias,
                           float* __restrict__ out) {
    extern __shared__ unsigned short smem[];

    const int tid  = threadIdx.x;
    const int lane = tid & 31;
    const int wave = tid >> 5;
    const int pl     = wave >> 2;          // pixel-in-tile 0..3
    const int quad   = wave & 3;           // output quadrant
    const int n_base = (quad & 1) * 64;
    const int o_base = (quad >> 1) * 64;

    const int p0 = blockIdx.x * PT;        // 256 blocks * 4 pixels
    const int h0 = p0 >> 5;                // output row (shared by all 4 pixels)
    const int w0 = p0 & 31;                // first output col

    // lane-varying row index (n for A, o for B); wave-uniform k-group (scalar!)
    const int lane7 = tid & 127;
    const int grp   = __builtin_amdgcn_readfirstlane(tid >> 7);   // 0..3

    // per-lane global offsets, computed once
    const int voffx = lane7 * (C_IN * HW * HW);   // x element offset for n=lane7
    const int vofff = lane7 * (RF * PIX);         // filt element offset for o=lane7

    // ---- staging helpers ----
    auto stageA = [&](int r0, unsigned short* lA) {
        // im2col gather from x (L2-resident). Thread covers n=lane7,
        // kk = 2*grp + 8*s (+1), all 4 pixels. 32 elems = 16 packed stores.
#pragma unroll
        for (int p = 0; p < 4; ++p) {
            const int wb = w0 + p - 1;
            unsigned* dst = (unsigned*)(lA + (p * 128 + lane7) * KSTR + 2 * grp);
            float va[8];
#pragma unroll
            for (int s = 0; s < 4; ++s) {
                const int r   = r0 + 2 * grp + 8 * s;    // scalar chain
                const int c0  = r / 9;
                const int rem = r - 9 * c0;
                const int i0  = rem / 3;
                const int j0  = rem - 3 * i0;
                int c1 = c0, i1 = i0, j1 = j0 + 1;
                if (j1 == 3) { j1 = 0; i1 = i0 + 1; if (i1 == 3) { i1 = 0; c1 = c0 + 1; } }
                const int hh0 = h0 + i0 - 1, hh1 = h0 + i1 - 1;
                const int ww0 = wb + j0,     ww1 = wb + j1;
                const bool ok0 = ((unsigned)hh0 < 32u) && ((unsigned)ww0 < 32u);
                const bool ok1 = ((unsigned)hh1 < 32u) && ((unsigned)ww1 < 32u);
                const int off0 = ok0 ? (c0 * 1024 + hh0 * 32 + ww0) : 0;  // scalar select
                const int off1 = ok1 ? (c1 * 1024 + hh1 * 32 + ww1) : 0;
                const float* b0 = x + off0;
                const float* b1 = x + off1;
                float v0 = b0[voffx];
                float v1 = b1[voffx];
                va[2 * s]     = ok0 ? v0 : 0.f;
                va[2 * s + 1] = ok1 ? v1 : 0.f;
            }
#pragma unroll
            for (int s = 0; s < 4; ++s)
                dst[4 * s] = pack2bf(va[2 * s], va[2 * s + 1]);   // 16B-stride imm offsets
        }
    };

    auto loadB = [&](int r0, float4* breg) {
        // filters[o=lane7, r0+kk(+1), p0..p0+3]; two 16B loads per it off one
        // scalar base (second at +PIX)
#pragma unroll
        for (int it = 0; it < 4; ++it) {
            const int kk = 2 * (grp + 4 * it);            // scalar
            const float* bp = filt + (size_t)(r0 + kk) * PIX + p0;
            breg[2 * it]     = *(const float4*)(bp + vofff);
            breg[2 * it + 1] = *(const float4*)(bp + vofff + PIX);
        }
    };

    auto storeB = [&](const float4* breg, unsigned short* lB) {
        unsigned* d0 = (unsigned*)(lB + (0 * 128 + lane7) * KSTR + 2 * grp);
        unsigned* d1 = (unsigned*)(lB + (1 * 128 + lane7) * KSTR + 2 * grp);
        unsigned* d2 = (unsigned*)(lB + (2 * 128 + lane7) * KSTR + 2 * grp);
        unsigned* d3 = (unsigned*)(lB + (3 * 128 + lane7) * KSTR + 2 * grp);
#pragma unroll
        for (int it = 0; it < 4; ++it) {
            const float4 f1 = breg[2 * it];
            const float4 f2 = breg[2 * it + 1];
            d0[4 * it] = pack2bf(f1.x, f2.x);
            d1[4 * it] = pack2bf(f1.y, f2.y);
            d2[4 * it] = pack2bf(f1.z, f2.z);
            d3[4 * it] = pack2bf(f1.w, f2.w);
        }
    };

    v8f acc[4][4];
#pragma unroll
    for (int mt = 0; mt < 4; ++mt)
#pragma unroll
        for (int nt = 0; nt < 4; ++nt)
            acc[mt][nt] = (v8f){0.f, 0.f, 0.f, 0.f, 0.f, 0.f, 0.f, 0.f};

    // ---- prologue: fill buffer 0 ----
    {
        float4 breg[8];
        loadB(0, breg);
        stageA(0, smem);
        storeB(breg, smem + PT * 128 * KSTR);
    }
    __syncthreads();

    const int m     = lane & 15;
    const int khalf = lane >> 4;

    for (int step = 0; step < NSTEP; ++step) {
        unsigned short* cur = smem + (step & 1) * BUFU;
        unsigned short* nxt = smem + ((step + 1) & 1) * BUFU;

        // issue next filter slab loads NOW (in flight during the WMMAs)
        float4 breg[8];
        if (step + 1 < NSTEP) loadB((step + 1) * KC, breg);

        // ---- Compute: 16 WMMAs per wave per K-step ----
        const unsigned short* Ab = cur + (pl * 128 + n_base) * KSTR;
        const unsigned short* Bb = cur + PT * 128 * KSTR + (pl * 128 + o_base) * KSTR;

        Frag bfr[4];
#pragma unroll
        for (int nt = 0; nt < 4; ++nt) {
            // B frag: lane column N = m, K in [khalf*16, khalf*16+16)
            const unsigned short* cp = Bb + (nt * 16 + m) * KSTR + khalf * 16;
            bfr[nt].q[0] = *(const uint4*)(cp);
            bfr[nt].q[1] = *(const uint4*)(cp + 8);
        }
#pragma unroll
        for (int mt = 0; mt < 4; ++mt) {
            // A frag: lane row M = m, K in [khalf*16, khalf*16+16)
            Frag af;
            const unsigned short* rp = Ab + (mt * 16 + m) * KSTR + khalf * 16;
            af.q[0] = *(const uint4*)(rp);
            af.q[1] = *(const uint4*)(rp + 8);
#pragma unroll
            for (int nt = 0; nt < 4; ++nt) {
                acc[mt][nt] = __builtin_amdgcn_wmma_f32_16x16x32_bf16(
                    false, af.v, false, bfr[nt].v, (short)0, acc[mt][nt],
                    false, false);
            }
        }

        // ---- stage next step into the other buffer ----
        if (step + 1 < NSTEP) {
            stageA((step + 1) * KC, nxt);
            storeB(breg, nxt + PT * 128 * KSTR);
        }
        __syncthreads();
    }

    // ---- Epilogue: stage through LDS, + bias, coalesced float4 stores ----
    // lout layout: [64 n][128 o][4 p] f32 = 128 KB (reuses staging LDS)
    float* lout = (float*)smem;
    const int ncol = lane & 15;
    const int mhi  = (lane >> 4) * 8;   // C/D: VGPR r holds M=r (lanes 0-15) / M=r+8 (lanes 16-31)

#pragma unroll
    for (int half = 0; half < 2; ++half) {
        if (n_base == half * 64) {
#pragma unroll
            for (int mt = 0; mt < 4; ++mt) {
                const int nl = mt * 16 + mhi;
#pragma unroll
                for (int nt = 0; nt < 4; ++nt) {
                    const int o = o_base + nt * 16 + ncol;
#pragma unroll
                    for (int r = 0; r < 8; ++r)
                        lout[((nl + r) * 128 + o) * 4 + pl] = acc[mt][nt][r];
                }
            }
        }
        __syncthreads();
        // cooperative write-out: 64n * 128o float4 (across p) = 8192, 16/thread
#pragma unroll 4
        for (int it = 0; it < 16; ++it) {
            const int idx = tid + it * 512;
            const int o   = idx & 127;
            const int nl  = idx >> 7;
            float4 v        = *(const float4*)&lout[(nl * 128 + o) * 4];
            const float4 bv = *(const float4*)(bias + (size_t)o * PIX + p0);
            v.x += bv.x; v.y += bv.y; v.z += bv.z; v.w += bv.w;
            *(float4*)(out + ((size_t)(half * 64 + nl) * C_OUT + o) * PIX + p0) = v;
        }
        __syncthreads();
    }
}

extern "C" void kernel_launch(void* const* d_in, const int* in_sizes, int n_in,
                              void* d_out, int out_size, void* d_ws, size_t ws_size,
                              hipStream_t stream) {
    (void)in_sizes; (void)n_in; (void)d_ws; (void)ws_size; (void)out_size;
    const float* x    = (const float*)d_in[0];
    const float* filt = (const float*)d_in[1];
    const float* bias = (const float*)d_in[2];
    float* out        = (float*)d_out;

    const int blocks = PIX / PT;                                  // 256
    size_t lds = (size_t)2 * BUFU * sizeof(unsigned short);       // 160 KB (double buffer)
    const size_t lds_epi = (size_t)64 * 128 * 4 * sizeof(float);  // 128 KB epilogue reuse
    if (lds_epi > lds) lds = lds_epi;
    localconv2d_wmma_bf16<<<blocks, 512, lds, stream>>>(x, filt, bias, out);
}
